// RNNModel_39805756899817
// MI455X (gfx1250) — matrix operand order repaired and checked by reading, not compiled, not run on hardware
//
#include <hip/hip_runtime.h>

typedef __bf16 bf16;
typedef __attribute__((ext_vector_type(16))) __bf16 v16bf;
typedef __attribute__((ext_vector_type(8)))  float  v8f;
typedef __attribute__((ext_vector_type(4)))  float  f4;

#define T_STEPS 512
#define FDIM    32
#define UDIM    8

// LDS layout (float indices)
#define H_BASE     0     // 16 rows x 8 floats = h transpose buffer
#define Z_BASE     128   // 8 floats, permanently zero (read by lanes 16..31)
#define DUMP_BASE  136   // write sink for lanes with (lane&15) >= 8
#define LDS_FLOATS 256

__device__ __forceinline__ float tanh_fast(float x) {
  // tanh(x) = 1 - 2/(e^{2x}+1); v_exp_f32 + v_rcp_f32
  float e = __expf(2.0f * x);
  return 1.0f - 2.0f * __builtin_amdgcn_rcpf(e + 1.0f);
}
__device__ __forceinline__ float sigmoid_fast(float x) {
  return __builtin_amdgcn_rcpf(1.0f + __expf(-x));
}
// single v_med3_num_f32, no canonicalize (vs select->fmaxnum's 2 ops)
__device__ __forceinline__ float reluf(float x) {
  return __builtin_amdgcn_fmed3f(x, 0.0f, __builtin_inff());
}
// xor-by-8 within each 16-lane row: v_permlane16_b32 (VALU, no DS round trip)
__device__ __forceinline__ float xor8(float v) {
  int r = __builtin_amdgcn_permlane16(
      __builtin_bit_cast(int, v), __builtin_bit_cast(int, v),
      0xFEDCBA98u, 0x76543210u, false, false);
  return __builtin_bit_cast(float, r);
}

// One wave (32 lanes) per 16 batch rows. Fused: input GEMM (WMMA bf16,
// fp32 accumulate) + recurrent GEMM (WMMA, K=8 zero-padded) + gates + MLP head.
__global__ __launch_bounds__(32)
void lstm_fused_kernel(const float* __restrict__ x,
                       const float* __restrict__ kernel,      // [32, 32]
                       const float* __restrict__ rec_kernel,  // [8, 32]
                       const float* __restrict__ bias,        // [32]
                       const float* __restrict__ w1,          // [8, 64]
                       const float* __restrict__ b1,          // [64]
                       const float* __restrict__ w2,          // [64, 4]
                       const float* __restrict__ b2,          // [4]
                       float* __restrict__ out)               // [B, 4]
{
  __shared__ float lds[LDS_FLOATS];
  const int lane   = threadIdx.x;      // 0..31 (wave32)
  const int half   = lane >> 4;        // 0 | 1
  const int nloc   = lane & 15;        // N within tile / local row id
  const int u      = lane & 7;
  const int rowBlk = blockIdx.x * 16;

  // zero h area + zero region (dump area may hold garbage, never read)
  for (int i = lane; i < LDS_FLOATS; i += 32) lds[i] = 0.0f;

  // ---- Preload B operands as 32x16 bf16 B, contiguous K order (same as A) ----
  // lanes 0-15: N=nloc, VGPR j holds K=2j,2j+1 (K 0..15); lanes 16-31: K 16..31.
  v16bf bk[2], br[2];
  #pragma unroll
  for (int t = 0; t < 2; ++t) {
    int n  = nloc + 16 * t;
    int kb = half * 16;
    #pragma unroll
    for (int j = 0; j < 8; ++j) {
      bk[t][2*j+0] = (bf16)kernel[(kb + 2*j + 0) * 32 + n];
      bk[t][2*j+1] = (bf16)kernel[(kb + 2*j + 1) * 32 + n];
      float r0 = 0.0f, r1 = 0.0f;          // rec_kernel zero-padded to K=32
      if (half == 0 && j < 4) {
        r0 = rec_kernel[(2*j + 0) * 32 + n];
        r1 = rec_kernel[(2*j + 1) * 32 + n];
      }
      br[t][2*j+0] = (bf16)r0;
      br[t][2*j+1] = (bf16)r1;
    }
  }

  // bias folded into the WMMA C operand (loop-invariant vectors, no per-step movs)
  v8f biasv0, biasv1;
  #pragma unroll
  for (int r = 0; r < 8; ++r) { biasv0[r] = bias[nloc]; biasv1[r] = bias[nloc + 16]; }

  // Branch-free LDS addressing (computed once):
  //  - h store: valid lanes ((lane&15)<8) -> h slot; others -> dump sink
  //  - A_h load: lanes 0-15 -> their h row; lanes 16-31 -> permanent zero region
  float* hstore = &lds[((lane & 15) < 8) ? (half * 64 + u) : (DUMP_BASE + lane)];
  const f4* hload = (const f4*)&lds[(lane < 16) ? (lane * UDIM) : Z_BASE];

  // A 16x32 bf16, contiguous K order used here: lanes 0-15 hold row M=lane, K 0..15;
  // lanes 16-31 hold row M=lane-16, K 16..31.
  const float* xrow = x + (size_t)(rowBlk + nloc) * (size_t)(T_STEPS * FDIM);
  const int c0 = half * 16;

  // Cell state: valid in lanes with (lane&15)<8; m = r + 8*half, u = lane&7.
  float creg[8];
  #pragma unroll
  for (int r = 0; r < 8; ++r) creg[r] = 0.0f;

  // Prime the x double buffer (t = 0): 4x float4 per lane, 2 KB/step/wave total.
  f4 xq[4];
  {
    const f4* p  = (const f4*)(xrow + c0);
    const f4* p2 = (const f4*)(xrow + c0 + 8);
    xq[0] = p[0]; xq[1] = p[1]; xq[2] = p2[0]; xq[3] = p2[1];
  }

  for (int t = 0; t < T_STEPS; ++t) {
    // Pack current x tile into A operand (bf16)
    v16bf ax;
    #pragma unroll
    for (int q = 0; q < 4; ++q)
      #pragma unroll
      for (int e = 0; e < 4; ++e) ax[4*q + e] = (bf16)xq[q][e];

    // Issue next timestep's loads now (register double buffer) + L2 prefetch t+2
    int tn = (t + 1 < T_STEPS) ? (t + 1) : t;
    {
      const f4* p  = (const f4*)(xrow + tn * FDIM + c0);
      const f4* p2 = (const f4*)(xrow + tn * FDIM + c0 + 8);
      xq[0] = p[0]; xq[1] = p[1]; xq[2] = p2[0]; xq[3] = p2[1];
      int tp = (t + 2 < T_STEPS) ? (t + 2) : tn;
      __builtin_prefetch(xrow + tp * FDIM + c0, 0, 1);
    }

    // Recurrent A operand from LDS-transposed h: K=0..7 live, rest zero.
    // Lanes 16-31 read the permanently-zero region -> no predication needed.
    f4 h0 = hload[0], h1 = hload[1];
    v16bf ah;
    #pragma unroll
    for (int e = 0; e < 4; ++e) {
      ah[e]      = (bf16)h0[e];
      ah[4 + e]  = (bf16)h1[e];
      ah[8 + e]  = (bf16)0.0f;   // K = 8..15 / 24..31 zero pad
      ah[12 + e] = (bf16)0.0f;
    }

    // z = bias + x_t @ kernel + h @ rec_kernel  (two N tiles, fp32 accumulate)
    v8f z0 = __builtin_amdgcn_wmma_f32_16x16x32_bf16(false, ax, false, bk[0], (short)0, biasv0, false, false);
    v8f z1 = __builtin_amdgcn_wmma_f32_16x16x32_bf16(false, ax, false, bk[1], (short)0, biasv1, false, false);
    z0 = __builtin_amdgcn_wmma_f32_16x16x32_bf16(false, ah, false, br[0], (short)0, z0, false, false);
    z1 = __builtin_amdgcn_wmma_f32_16x16x32_bf16(false, ah, false, br[1], (short)0, z1, false, false);

    // Gates: tile0 lanes {u, u+8} = {i, f}; tile1 lanes {u, u+8} = {g, o}.
    #pragma unroll
    for (int r = 0; r < 8; ++r) {
      float zi = z0[r];
      float zg = z1[r];
      float zf = xor8(zi);
      float zo = xor8(zg);
      float ig = reluf(zi);                // recurrent_activation = relu
      float fg = reluf(zf);
      float og = reluf(zo);
      float gg = tanh_fast(zg);            // activation = tanh
      float cn = fg * creg[r] + ig * gg;
      creg[r]  = cn;
      hstore[r * UDIM] = og * tanh_fast(cn);   // transpose for next A (or dump)
    }
  }

  __syncthreads();   // single-wave WG: cheap; orders last LDS stores

  // ---- MLP head: sigmoid(relu(h @ w1 + b1) @ w2 + b2), one lane per row ----
  if (lane < 16) {
    float hr[8];
    #pragma unroll
    for (int uu = 0; uu < 8; ++uu) hr[uu] = lds[lane * UDIM + uu];
    float o0 = b2[0], o1 = b2[1], o2 = b2[2], o3 = b2[3];
    for (int j = 0; j < 64; ++j) {
      float a = b1[j];
      #pragma unroll
      for (int uu = 0; uu < 8; ++uu) a = fmaf(hr[uu], w1[uu * 64 + j], a);
      a = reluf(a);
      o0 = fmaf(a, w2[j * 4 + 0], o0);
      o1 = fmaf(a, w2[j * 4 + 1], o1);
      o2 = fmaf(a, w2[j * 4 + 2], o2);
      o3 = fmaf(a, w2[j * 4 + 3], o3);
    }
    f4 res;
    res[0] = sigmoid_fast(o0);
    res[1] = sigmoid_fast(o1);
    res[2] = sigmoid_fast(o2);
    res[3] = sigmoid_fast(o3);
    *(f4*)(out + (size_t)(rowBlk + lane) * 4) = res;
  }
}

extern "C" void kernel_launch(void* const* d_in, const int* in_sizes, int n_in,
                              void* d_out, int out_size, void* d_ws, size_t ws_size,
                              hipStream_t stream) {
  const float* x   = (const float*)d_in[0];
  const float* krn = (const float*)d_in[1];
  const float* rk  = (const float*)d_in[2];
  const float* bs  = (const float*)d_in[3];
  const float* w1  = (const float*)d_in[4];
  const float* b1  = (const float*)d_in[5];
  const float* w2  = (const float*)d_in[6];
  const float* b2  = (const float*)d_in[7];
  float* out = (float*)d_out;

  const int B = in_sizes[0] / (T_STEPS * FDIM);   // 4096
  dim3 grid(B / 16), block(32);
  hipLaunchKernelGGL(lstm_fused_kernel, grid, block, 0, stream,
                     x, krn, rk, bs, w1, b1, w2, b2, out);
}
